// ContrastiveLoss_16587163697293
// MI455X (gfx1250) — matrix operand order repaired and checked by reading, not compiled, not run on hardware
//
#include <hip/hip_runtime.h>
#include <hip/hip_bf16.h>
#include <math.h>

// Problem constants (from reference): B=8, M=256, D=768, K=100 -> N=2048
#define NROWS 2048
#define DIM   768
#define KNEG  100

typedef __attribute__((ext_vector_type(16))) __bf16 v16bf;
typedef __attribute__((ext_vector_type(8)))  __bf16 v8bf;
typedef __attribute__((ext_vector_type(8)))  float  v8f;

static __device__ __forceinline__ unsigned short f2bf(float f) {
    union { float f; unsigned u; } v; v.f = f;
    unsigned u = v.u;
    // round-to-nearest-even truncation to bf16
    unsigned r = u + 0x7FFFu + ((u >> 16) & 1u);
    return (unsigned short)(r >> 16);
}

// ---------------------------------------------------------------------------
// Kernel 1: L2-normalize each row of preds and targets, emit bf16 operands.
// One wave (32 lanes) per row; 768/32 = 24 elems per lane. 8 rows per block.
// ---------------------------------------------------------------------------
__global__ __launch_bounds__(256) void normalize_rows_kernel(
    const float* __restrict__ preds, const float* __restrict__ targets,
    unsigned short* __restrict__ pnb, unsigned short* __restrict__ tnb) {
    int row  = blockIdx.x * 8 + (threadIdx.x >> 5);
    int lane = threadIdx.x & 31;
    const float* src;
    unsigned short* dst;
    if (row < NROWS) { src = preds   + (size_t)row * DIM;           dst = pnb + (size_t)row * DIM; }
    else             { src = targets + (size_t)(row - NROWS) * DIM; dst = tnb + (size_t)(row - NROWS) * DIM; }

    float vals[24];
    float ss = 0.0f;
#pragma unroll
    for (int j = 0; j < 24; ++j) {
        float v = src[lane + 32 * j];
        vals[j] = v;
        ss += v * v;
    }
    // wave32 reduction
#pragma unroll
    for (int off = 16; off >= 1; off >>= 1)
        ss += __shfl_xor(ss, off, 32);
    float inv = 1.0f / fmaxf(sqrtf(ss), 1e-8f);
#pragma unroll
    for (int j = 0; j < 24; ++j)
        dst[lane + 32 * j] = f2bf(vals[j] * inv);
}

// ---------------------------------------------------------------------------
// Kernel 2: zero the scalar output (harness poisons d_out).
// ---------------------------------------------------------------------------
__global__ void zero_out_kernel(float* out) { out[0] = 0.0f; }

// ---------------------------------------------------------------------------
// Kernel 3 (fused): per 16-row block, compute the full 16x2048 sim strip with
// v_wmma_f32_16x16x32_bf16 into 128 KB of LDS, then gather pos + 100 negatives
// per row from LDS, log-softmax-reduce with wave32 shuffles, atomicAdd mean.
// Block = 256 threads = 8 waves; wave w owns column tiles [16w, 16w+16).
// ---------------------------------------------------------------------------
__global__ __launch_bounds__(256) void contrastive_fused_kernel(
    const unsigned short* __restrict__ pnb_, const unsigned short* __restrict__ tnb_,
    const int* __restrict__ negidx, float* __restrict__ out) {
    extern __shared__ float simLDS[];  // 16 rows x 2048 cols = 131072 bytes

    const __bf16* pnb = (const __bf16*)pnb_;
    const __bf16* tnb = (const __bf16*)tnb_;

    const int r0   = blockIdx.x * 16;
    const int wave = threadIdx.x >> 5;
    const int lane = threadIdx.x & 31;
    const int m    = lane & 15;   // A row / B column within tile
    const int kh   = lane >> 4;   // K-half select per ISA 16-bit layouts

    // ---- Phase 1: WMMA GEMM strip (uniform control flow, EXEC all ones) ----
    const __bf16* arow = pnb + (size_t)(r0 + m) * DIM + 8 * kh;
    for (int j = 0; j < 16; ++j) {
        const int c0 = (wave * 16 + j) * 16;
        const __bf16* brow = tnb + (size_t)(c0 + m) * DIM + 8 * kh;
        v8f acc = {};
#pragma unroll
        for (int kt = 0; kt < 24; ++kt) {
            const int kb = kt * 32;
            // A fragment: lane holds M=m; elems 0..7 -> K = kb+8*kh+e,
            //             elems 8..15 -> K = kb+16+8*kh+(e-8)   (ISA 7.12.2)
            v8bf alo = *(const v8bf*)(arow + kb);
            v8bf ahi = *(const v8bf*)(arow + kb + 16);
            v16bf a  = __builtin_shufflevector(alo, ahi,
                          0,1,2,3,4,5,6,7,8,9,10,11,12,13,14,15);
            // B fragment: lane holds N=m; elems 0..7 -> K = kb+8*kh+e,
            //             elems 8..15 -> K = kb+16+8*kh+(e-8) (same as A)
            v8bf blo = *(const v8bf*)(brow + kb);
            v8bf bhi = *(const v8bf*)(brow + kb + 16);
            v16bf b  = __builtin_shufflevector(blo, bhi,
                          0,1,2,3,4,5,6,7,8,9,10,11,12,13,14,15);
            acc = __builtin_amdgcn_wmma_f32_16x16x32_bf16(
                      false, a, false, b, (short)0, acc, false, false);
        }
        // C/D layout: VGPR i -> M = i + 8*kh, N = lane&15
#pragma unroll
        for (int i = 0; i < 8; ++i)
            simLDS[(i + 8 * kh) * NROWS + c0 + m] = acc[i];
    }

    __syncthreads();

    // ---- Phase 2: gather + log-softmax; wave w handles local rows 2w, 2w+1 --
    for (int rr = 0; rr < 2; ++rr) {
        const int ml = wave * 2 + rr;       // local row 0..15
        const int gi = r0 + ml;             // global row
        const float* srow = simLDS + (size_t)ml * NROWS;

        const float l0 = srow[gi] * 10.0f;  // pos logit (1/TEMPERATURE)
        float lt[4];
        int   cnt = 0;
        float mx  = l0;
        for (int t = lane; t < KNEG; t += 32) {
            int   idx = negidx[(size_t)gi * KNEG + t];
            float v   = srow[idx] * 10.0f;
            lt[cnt++] = v;
            mx = fmaxf(mx, v);
        }
#pragma unroll
        for (int off = 16; off >= 1; off >>= 1)
            mx = fmaxf(mx, __shfl_xor(mx, off, 32));
        float se = 0.0f;
        for (int q = 0; q < cnt; ++q)
            se += __expf(lt[q] - mx);
#pragma unroll
        for (int off = 16; off >= 1; off >>= 1)
            se += __shfl_xor(se, off, 32);
        se += __expf(l0 - mx);              // pos term, uniform across lanes
        const float loss = -(l0 - mx - __logf(se));
        if (lane == 0)
            atomicAdd(out, loss * (1.0f / (float)NROWS));
    }
}

// ---------------------------------------------------------------------------
extern "C" void kernel_launch(void* const* d_in, const int* in_sizes, int n_in,
                              void* d_out, int out_size, void* d_ws, size_t ws_size,
                              hipStream_t stream) {
    (void)in_sizes; (void)n_in; (void)out_size; (void)ws_size;
    const float* preds   = (const float*)d_in[0];
    const float* targets = (const float*)d_in[1];
    const int*   negidx  = (const int*)d_in[2];
    float*       out     = (float*)d_out;

    unsigned short* pnb = (unsigned short*)d_ws;                   // 2048*768 bf16
    unsigned short* tnb = pnb + (size_t)NROWS * DIM;               // 2048*768 bf16

    const int ldsBytes = 16 * NROWS * (int)sizeof(float);          // 128 KB (CDNA5 WGP: 320 KB)
    hipFuncSetAttribute((const void*)contrastive_fused_kernel,
                        hipFuncAttributeMaxDynamicSharedMemorySize, ldsBytes);

    normalize_rows_kernel<<<(2 * NROWS) / 8, 256, 0, stream>>>(preds, targets, pnb, tnb);
    zero_out_kernel<<<1, 1, 0, stream>>>(out);
    contrastive_fused_kernel<<<NROWS / 16, 256, ldsBytes, stream>>>(pnb, tnb, negidx, out);
}
